// MinGRUBlock_44547400794170
// MI455X (gfx1250) — matrix operand order repaired and checked by reading, not compiled, not run on hardware
//
#include <hip/hip_runtime.h>

// MinGRU block for MI455X (gfx1250): bf16 WMMA GEMMs + fp32 chunked scan.
// All matrix math via v_wmma_f32_16x16x32_bf16 (wave32).

#define B_   4
#define L_   8192
#define D_   1024
#define M_   (B_ * L_)      // 32768
#define CH_  128            // scan chunk length
#define C_   (L_ / CH_)     // 64 chunks per sequence

typedef __attribute__((ext_vector_type(16))) __bf16 v16bf;
typedef __attribute__((ext_vector_type(8)))  __bf16 v8bf;
typedef __attribute__((ext_vector_type(8)))  float  v8f;

#define LDSPAD 40   // padded row stride (bf16 elems) for 32-wide K tiles

__device__ __forceinline__ __bf16 f2bf(float f) {
  unsigned u = __builtin_bit_cast(unsigned, f);
  unsigned r = u + 0x7FFFu + ((u >> 16) & 1u);        // round-to-nearest-even
  return __builtin_bit_cast(__bf16, (unsigned short)(r >> 16));
}

__device__ __forceinline__ float sigm(float z) { return 1.0f / (1.0f + __expf(-z)); }

// A-matrix 16x32 bf16 fragment: lane holds row M=lane%16;
// elems 0..7 = K[k0..k0+7], elems 8..15 = K[k0+16..k0+23], k0 = (lane>=16)?8:0.
__device__ __forceinline__ v16bf load_frag_a(const __bf16* t, int rowBase, int lane) {
  int r  = rowBase + (lane & 15);
  int k0 = (lane >> 4) * 8;
  union { v16bf v; v8bf h[2]; } u;
  u.h[0] = *(const v8bf*)(t + r * LDSPAD + k0);
  u.h[1] = *(const v8bf*)(t + r * LDSPAD + k0 + 16);
  return u.v;
}

// B-matrix 32x16 bf16 fragment (tile stored as W rows = N x K):
// lane holds col N=lane%16; elems 0..7 = K[k0..k0+7], elems 8..15 = K[k0+16..k0+23], k0 = (lane>=16)?8:0.
__device__ __forceinline__ v16bf load_frag_b(const __bf16* t, int rowBase, int lane) {
  int r  = rowBase + (lane & 15);
  int k0 = (lane >> 4) * 8;
  union { v16bf v; v8bf h[2]; } u;
  u.h[0] = *(const v8bf*)(t + r * LDSPAD + k0);
  u.h[1] = *(const v8bf*)(t + r * LDSPAD + k0 + 16);
  return u.v;
}

// ---------------- weight conversion ----------------
__global__ void cvt_weights_kernel(const float* __restrict__ a, const float* __restrict__ b,
                                   const float* __restrict__ c, const float* __restrict__ d,
                                   const float* __restrict__ e,
                                   __bf16* __restrict__ oa, __bf16* __restrict__ ob,
                                   __bf16* __restrict__ oc, __bf16* __restrict__ od,
                                   __bf16* __restrict__ oe) {
  int i = blockIdx.x * 256 + threadIdx.x;
  if (i < D_ * D_) {
    oa[i] = f2bf(a[i]); ob[i] = f2bf(b[i]); oc[i] = f2bf(c[i]);
    od[i] = f2bf(d[i]); oe[i] = f2bf(e[i]);
  }
}

// ---------------- RMSNorm -> bf16 ----------------
__global__ __launch_bounds__(256) void rmsnorm_bf16_kernel(const float* __restrict__ x,
                                                           const float* __restrict__ w,
                                                           __bf16* __restrict__ y) {
  const size_t row = blockIdx.x;
  const float* xr = x + row * D_;
  __shared__ float red[8];
  float v[4];
  float s = 0.f;
#pragma unroll
  for (int i = 0; i < 4; ++i) { v[i] = xr[threadIdx.x + i * 256]; s += v[i] * v[i]; }
#pragma unroll
  for (int off = 16; off > 0; off >>= 1) s += __shfl_xor(s, off, 32);
  if ((threadIdx.x & 31) == 0) red[threadIdx.x >> 5] = s;
  __syncthreads();
  float tot = 0.f;
#pragma unroll
  for (int i = 0; i < 8; ++i) tot += red[i];
  float scale = rsqrtf(tot * (1.0f / D_) + 1e-6f);
#pragma unroll
  for (int i = 0; i < 4; ++i) {
    int dcol = threadIdx.x + i * 256;
    y[row * D_ + dcol] = f2bf(v[i] * scale * w[dcol]);
  }
}

// ---------------- dual GEMM + gate epilogue ----------------
// yg = A@Wg^T, yc = A@Wc^T ; beta = sigmoid(yg); a = 1-beta; x = beta*yc
__global__ __launch_bounds__(256, 2) void gemm_dual_gate_kernel(
    const __bf16* __restrict__ A, const __bf16* __restrict__ Wg, const __bf16* __restrict__ Wc,
    float* __restrict__ a_out, float* __restrict__ x_out) {
  constexpr int K = D_;
  __shared__ alignas(16) __bf16 sA[128 * LDSPAD];
  __shared__ alignas(16) __bf16 sW[2 * 64 * LDSPAD];
  const int tid = threadIdx.x;
  const int lane = tid & 31, wave = tid >> 5;
  const int wm = wave & 3, wn = wave >> 2;
  const size_t rowM0 = (size_t)blockIdx.x * 128;
  const int colN0 = blockIdx.y * 64;

  v8f accg[2][2], accc[2][2];
#pragma unroll
  for (int i = 0; i < 2; ++i)
#pragma unroll
    for (int j = 0; j < 2; ++j) { v8f z = {}; accg[i][j] = z; accc[i][j] = z; }

  for (int k0 = 0; k0 < K; k0 += 32) {
    __syncthreads();
#pragma unroll
    for (int cc = 0; cc < 2; ++cc) {
      int c = tid + cc * 256;                 // 512 chunks of 8 bf16 each
      int r = c >> 2, col = (c & 3) * 8;
      *(v8bf*)(sA + r * LDSPAD + col) = *(const v8bf*)(A + (rowM0 + r) * K + k0 + col);
      int mat = c >> 8, rw = (c >> 2) & 63;
      const __bf16* src = mat ? Wc : Wg;
      *(v8bf*)(sW + (mat * 64 + rw) * LDSPAD + col) =
          *(const v8bf*)(src + (size_t)(colN0 + rw) * K + k0 + col);
    }
    if (k0 + 32 < K)
      __builtin_prefetch(A + (rowM0 + (tid >> 1)) * K + k0 + 32, 0, 1);
    __syncthreads();

    v16bf af0 = load_frag_a(sA, wm * 32, lane);
    v16bf af1 = load_frag_a(sA, wm * 32 + 16, lane);
    v16bf bg0 = load_frag_b(sW, wn * 32, lane);
    v16bf bg1 = load_frag_b(sW, wn * 32 + 16, lane);
    v16bf bc0 = load_frag_b(sW + 64 * LDSPAD, wn * 32, lane);
    v16bf bc1 = load_frag_b(sW + 64 * LDSPAD, wn * 32 + 16, lane);

    accg[0][0] = __builtin_amdgcn_wmma_f32_16x16x32_bf16(false, af0, false, bg0, (short)0, accg[0][0], false, false);
    accg[0][1] = __builtin_amdgcn_wmma_f32_16x16x32_bf16(false, af0, false, bg1, (short)0, accg[0][1], false, false);
    accg[1][0] = __builtin_amdgcn_wmma_f32_16x16x32_bf16(false, af1, false, bg0, (short)0, accg[1][0], false, false);
    accg[1][1] = __builtin_amdgcn_wmma_f32_16x16x32_bf16(false, af1, false, bg1, (short)0, accg[1][1], false, false);
    accc[0][0] = __builtin_amdgcn_wmma_f32_16x16x32_bf16(false, af0, false, bc0, (short)0, accc[0][0], false, false);
    accc[0][1] = __builtin_amdgcn_wmma_f32_16x16x32_bf16(false, af0, false, bc1, (short)0, accc[0][1], false, false);
    accc[1][0] = __builtin_amdgcn_wmma_f32_16x16x32_bf16(false, af1, false, bc0, (short)0, accc[1][0], false, false);
    accc[1][1] = __builtin_amdgcn_wmma_f32_16x16x32_bf16(false, af1, false, bc1, (short)0, accc[1][1], false, false);
  }

  const int mloc = (lane >> 4) * 8, nl = lane & 15;
#pragma unroll
  for (int i = 0; i < 2; ++i)
#pragma unroll
    for (int j = 0; j < 2; ++j) {
      size_t mbase = rowM0 + wm * 32 + i * 16 + mloc;
      size_t ncol = (size_t)colN0 + wn * 32 + j * 16 + nl;
#pragma unroll
      for (int r = 0; r < 8; ++r) {
        float beta = sigm(accg[i][j][r]);
        size_t idx = (mbase + r) * D_ + ncol;
        a_out[idx] = 1.0f - beta;
        x_out[idx] = beta * accc[i][j][r];
      }
    }
}

// ---------------- dual GEMM + SwiGLU epilogue ----------------
__global__ __launch_bounds__(256, 2) void gemm_dual_swiglu_kernel(
    const __bf16* __restrict__ A, const __bf16* __restrict__ W1, const __bf16* __restrict__ W3,
    __bf16* __restrict__ g_out) {
  constexpr int K = D_;
  __shared__ alignas(16) __bf16 sA[128 * LDSPAD];
  __shared__ alignas(16) __bf16 sW[2 * 64 * LDSPAD];
  const int tid = threadIdx.x;
  const int lane = tid & 31, wave = tid >> 5;
  const int wm = wave & 3, wn = wave >> 2;
  const size_t rowM0 = (size_t)blockIdx.x * 128;
  const int colN0 = blockIdx.y * 64;

  v8f a1[2][2], a3[2][2];
#pragma unroll
  for (int i = 0; i < 2; ++i)
#pragma unroll
    for (int j = 0; j < 2; ++j) { v8f z = {}; a1[i][j] = z; a3[i][j] = z; }

  for (int k0 = 0; k0 < K; k0 += 32) {
    __syncthreads();
#pragma unroll
    for (int cc = 0; cc < 2; ++cc) {
      int c = tid + cc * 256;
      int r = c >> 2, col = (c & 3) * 8;
      *(v8bf*)(sA + r * LDSPAD + col) = *(const v8bf*)(A + (rowM0 + r) * K + k0 + col);
      int mat = c >> 8, rw = (c >> 2) & 63;
      const __bf16* src = mat ? W3 : W1;
      *(v8bf*)(sW + (mat * 64 + rw) * LDSPAD + col) =
          *(const v8bf*)(src + (size_t)(colN0 + rw) * K + k0 + col);
    }
    if (k0 + 32 < K)
      __builtin_prefetch(A + (rowM0 + (tid >> 1)) * K + k0 + 32, 0, 1);
    __syncthreads();

    v16bf af0 = load_frag_a(sA, wm * 32, lane);
    v16bf af1 = load_frag_a(sA, wm * 32 + 16, lane);
    v16bf b10 = load_frag_b(sW, wn * 32, lane);
    v16bf b11 = load_frag_b(sW, wn * 32 + 16, lane);
    v16bf b30 = load_frag_b(sW + 64 * LDSPAD, wn * 32, lane);
    v16bf b31 = load_frag_b(sW + 64 * LDSPAD, wn * 32 + 16, lane);

    a1[0][0] = __builtin_amdgcn_wmma_f32_16x16x32_bf16(false, af0, false, b10, (short)0, a1[0][0], false, false);
    a1[0][1] = __builtin_amdgcn_wmma_f32_16x16x32_bf16(false, af0, false, b11, (short)0, a1[0][1], false, false);
    a1[1][0] = __builtin_amdgcn_wmma_f32_16x16x32_bf16(false, af1, false, b10, (short)0, a1[1][0], false, false);
    a1[1][1] = __builtin_amdgcn_wmma_f32_16x16x32_bf16(false, af1, false, b11, (short)0, a1[1][1], false, false);
    a3[0][0] = __builtin_amdgcn_wmma_f32_16x16x32_bf16(false, af0, false, b30, (short)0, a3[0][0], false, false);
    a3[0][1] = __builtin_amdgcn_wmma_f32_16x16x32_bf16(false, af0, false, b31, (short)0, a3[0][1], false, false);
    a3[1][0] = __builtin_amdgcn_wmma_f32_16x16x32_bf16(false, af1, false, b30, (short)0, a3[1][0], false, false);
    a3[1][1] = __builtin_amdgcn_wmma_f32_16x16x32_bf16(false, af1, false, b31, (short)0, a3[1][1], false, false);
  }

  const int mloc = (lane >> 4) * 8, nl = lane & 15;
#pragma unroll
  for (int i = 0; i < 2; ++i)
#pragma unroll
    for (int j = 0; j < 2; ++j) {
      size_t mbase = rowM0 + wm * 32 + i * 16 + mloc;
      size_t ncol = (size_t)colN0 + wn * 32 + j * 16 + nl;
#pragma unroll
      for (int r = 0; r < 8; ++r) {
        float t1 = a1[i][j][r], t3 = a3[i][j][r];
        g_out[(mbase + r) * D_ + ncol] = f2bf(t1 * sigm(t1) * t3);
      }
    }
}

// ---------------- GEMM + residual add ----------------
__global__ __launch_bounds__(256, 2) void gemm_add_kernel(
    const __bf16* __restrict__ G, const __bf16* __restrict__ W2,
    const float* __restrict__ h, float* __restrict__ out) {
  constexpr int K = D_;
  __shared__ alignas(16) __bf16 sA[128 * LDSPAD];
  __shared__ alignas(16) __bf16 sW[64 * LDSPAD];
  const int tid = threadIdx.x;
  const int lane = tid & 31, wave = tid >> 5;
  const int wm = wave & 3, wn = wave >> 2;
  const size_t rowM0 = (size_t)blockIdx.x * 128;
  const int colN0 = blockIdx.y * 64;

  v8f acc[2][2];
#pragma unroll
  for (int i = 0; i < 2; ++i)
#pragma unroll
    for (int j = 0; j < 2; ++j) { v8f z = {}; acc[i][j] = z; }

  for (int k0 = 0; k0 < K; k0 += 32) {
    __syncthreads();
#pragma unroll
    for (int cc = 0; cc < 2; ++cc) {
      int c = tid + cc * 256;
      int r = c >> 2, col = (c & 3) * 8;
      *(v8bf*)(sA + r * LDSPAD + col) = *(const v8bf*)(G + (rowM0 + r) * K + k0 + col);
    }
    {
      int c = tid;                             // 256 chunks for 64x32 W tile
      int rw = c >> 2, col = (c & 3) * 8;
      *(v8bf*)(sW + rw * LDSPAD + col) =
          *(const v8bf*)(W2 + (size_t)(colN0 + rw) * K + k0 + col);
    }
    if (k0 + 32 < K)
      __builtin_prefetch(G + (rowM0 + (tid >> 1)) * K + k0 + 32, 0, 1);
    __syncthreads();

    v16bf af0 = load_frag_a(sA, wm * 32, lane);
    v16bf af1 = load_frag_a(sA, wm * 32 + 16, lane);
    v16bf b0  = load_frag_b(sW, wn * 32, lane);
    v16bf b1  = load_frag_b(sW, wn * 32 + 16, lane);

    acc[0][0] = __builtin_amdgcn_wmma_f32_16x16x32_bf16(false, af0, false, b0, (short)0, acc[0][0], false, false);
    acc[0][1] = __builtin_amdgcn_wmma_f32_16x16x32_bf16(false, af0, false, b1, (short)0, acc[0][1], false, false);
    acc[1][0] = __builtin_amdgcn_wmma_f32_16x16x32_bf16(false, af1, false, b0, (short)0, acc[1][0], false, false);
    acc[1][1] = __builtin_amdgcn_wmma_f32_16x16x32_bf16(false, af1, false, b1, (short)0, acc[1][1], false, false);
  }

  const int mloc = (lane >> 4) * 8, nl = lane & 15;
#pragma unroll
  for (int i = 0; i < 2; ++i)
#pragma unroll
    for (int j = 0; j < 2; ++j) {
      size_t mbase = rowM0 + wm * 32 + i * 16 + mloc;
      size_t ncol = (size_t)colN0 + wn * 32 + j * 16 + nl;
#pragma unroll
      for (int r = 0; r < 8; ++r) {
        size_t idx = (mbase + r) * D_ + ncol;
        out[idx] = h[idx] + acc[i][j][r];
      }
    }
}

// ---------------- chunked linear-recurrence scan ----------------
__global__ void scan_pass1_kernel(const float* __restrict__ a, const float* __restrict__ x,
                                  float* __restrict__ chA, float* __restrict__ chX) {
  int t = blockIdx.x * 256 + threadIdx.x;     // 0 .. B*C*D-1
  int d = t & (D_ - 1);
  int chunk = (t >> 10) & (C_ - 1);
  int b = t >> 16;
  size_t base = ((size_t)b * L_ + (size_t)chunk * CH_) * D_ + d;
  float A = 1.f, X = 0.f;
#pragma unroll 4
  for (int i = 0; i < CH_; ++i) {
    float at = a[base + (size_t)i * D_];
    float xt = x[base + (size_t)i * D_];
    A *= at;
    X = at * X + xt;
  }
  chA[t] = A;
  chX[t] = X;
}

__global__ void scan_pass2_kernel(const float* __restrict__ chA, const float* __restrict__ chX,
                                  float* __restrict__ carry) {
  int t = blockIdx.x * 256 + threadIdx.x;     // 0 .. B*D-1
  int d = t & (D_ - 1);
  int b = t >> 10;
  float h = 0.f;
  for (int c = 0; c < C_; ++c) {
    size_t i = ((size_t)b * C_ + c) * D_ + d;
    carry[i] = h;
    h = chA[i] * h + chX[i];
  }
}

__global__ void scan_pass3_kernel(const float* __restrict__ a, const float* __restrict__ carry,
                                  float* __restrict__ hx) {
  int t = blockIdx.x * 256 + threadIdx.x;
  int d = t & (D_ - 1);
  int chunk = (t >> 10) & (C_ - 1);
  int b = t >> 16;
  size_t base = ((size_t)b * L_ + (size_t)chunk * CH_) * D_ + d;
  float h = carry[t];
#pragma unroll 4
  for (int i = 0; i < CH_; ++i) {
    float at = a[base + (size_t)i * D_];
    float xt = hx[base + (size_t)i * D_];
    h = at * h + xt;
    hx[base + (size_t)i * D_] = h;            // overwrite x with h in place
  }
}

extern "C" void kernel_launch(void* const* d_in, const int* in_sizes, int n_in,
                              void* d_out, int out_size, void* d_ws, size_t ws_size,
                              hipStream_t stream) {
  const float* inp  = (const float*)d_in[0];
  const float* Wg   = (const float*)d_in[1];
  const float* Wc   = (const float*)d_in[2];
  const float* w1   = (const float*)d_in[3];
  const float* w2   = (const float*)d_in[4];
  const float* w3   = (const float*)d_in[5];
  const float* lnw  = (const float*)d_in[6];
  const float* ffnw = (const float*)d_in[7];

  float* out = (float*)d_out;                 // out   : [B,L,D]
  float* hx  = out + (size_t)M_ * D_;         // hx    : [B,L,D] (second output)

  char* base = (char*)d_ws;
  size_t off = 0;
  auto take = [&](size_t bytes) -> char* {
    char* q = base + off;
    off += (bytes + 255) & ~(size_t)255;
    return q;
  };
  __bf16* wg_b  = (__bf16*)take((size_t)D_ * D_ * 2);
  __bf16* wc_b  = (__bf16*)take((size_t)D_ * D_ * 2);
  __bf16* w1_b  = (__bf16*)take((size_t)D_ * D_ * 2);
  __bf16* w2_b  = (__bf16*)take((size_t)D_ * D_ * 2);
  __bf16* w3_b  = (__bf16*)take((size_t)D_ * D_ * 2);
  __bf16* nrm_b = (__bf16*)take((size_t)M_ * D_ * 2);
  float*  a_f   = (float*) take((size_t)M_ * D_ * 4);
  __bf16* hn_b  = (__bf16*)take((size_t)M_ * D_ * 2);
  __bf16* g_b   = (__bf16*)take((size_t)M_ * D_ * 2);
  float*  chA   = (float*) take((size_t)B_ * C_ * D_ * 4);
  float*  chX   = (float*) take((size_t)B_ * C_ * D_ * 4);
  float*  carry = (float*) take((size_t)B_ * C_ * D_ * 4);

  dim3 gemmGrid(M_ / 128, D_ / 64);

  cvt_weights_kernel<<<(D_ * D_ + 255) / 256, 256, 0, stream>>>(
      Wg, Wc, w1, w2, w3, wg_b, wc_b, w1_b, w2_b, w3_b);
  rmsnorm_bf16_kernel<<<M_, 256, 0, stream>>>(inp, lnw, nrm_b);
  gemm_dual_gate_kernel<<<gemmGrid, 256, 0, stream>>>(nrm_b, wg_b, wc_b, a_f, hx);
  scan_pass1_kernel<<<(B_ * C_ * D_) / 256, 256, 0, stream>>>(a_f, hx, chA, chX);
  scan_pass2_kernel<<<(B_ * D_) / 256, 256, 0, stream>>>(chA, chX, carry);
  scan_pass3_kernel<<<(B_ * C_ * D_) / 256, 256, 0, stream>>>(a_f, carry, hx);
  rmsnorm_bf16_kernel<<<M_, 256, 0, stream>>>(hx, ffnw, hn_b);
  gemm_dual_swiglu_kernel<<<gemmGrid, 256, 0, stream>>>(hn_b, w1_b, w3_b, g_b);
  gemm_add_kernel<<<gemmGrid, 256, 0, stream>>>(g_b, w2_b, hx, out);
}
